// Attention_rope_slide_26577257627795
// MI455X (gfx1250) — matrix operand order repaired and checked by reading, not compiled, not run on hardware
//
#include <hip/hip_runtime.h>
#include <hip/hip_bf16.h>
#include <math.h>

typedef _Float16 h8   __attribute__((ext_vector_type(8)));
typedef _Float16 v16h __attribute__((ext_vector_type(16)));
typedef float    v8f  __attribute__((ext_vector_type(8)));
typedef unsigned int u32x4 __attribute__((ext_vector_type(4)));
typedef int          i32x8 __attribute__((ext_vector_type(8)));
typedef int          i32x4 __attribute__((ext_vector_type(4)));

#define GK 768
#define GN 2304

// ---------------------------------------------------------------------------
// helpers
// ---------------------------------------------------------------------------
__device__ __forceinline__ v16h ld16(const _Float16* p0, const _Float16* p1) {
    h8 a = *(const h8*)p0;
    h8 b = *(const h8*)p1;
    v16h r;
#pragma unroll
    for (int i = 0; i < 8; ++i) { r[i] = a[i]; r[i + 8] = b[i]; }
    return r;
}

__device__ __forceinline__ v8f wmma_f16(v16h a, v16h b, v8f c) {
    // D = A(16x32 f16) * B(32x16 f16) + C(16x16 f32)
    return __builtin_amdgcn_wmma_f32_16x16x32_f16(
        /*neg_a=*/false, a, /*neg_b=*/false, b,
        /*c_mod=*/(short)0, c, /*reuse_a=*/false, /*reuse_b=*/false);
}

// TDM: load a 2D tile (tile_x x tile_y f16 elements) from global (row stride
// stride_x elements) into LDS at lds_off, rows packed contiguously
// (row stride = tile_x elements). D# layout per CDNA5 ISA ch.8.
__device__ __forceinline__ void tdm_load_2d(const _Float16* gptr, unsigned lds_off,
                                            unsigned tile_x, unsigned tile_y,
                                            unsigned tensor_x, unsigned tensor_y,
                                            unsigned stride_x) {
    unsigned long long ga = (unsigned long long)(const void*)gptr;
    u32x4 g0;
    g0[0] = 1u;                                              // count=1, no gather
    g0[1] = lds_off;                                         // LDS byte address
    g0[2] = (unsigned)(ga & 0xFFFFFFFFu);                    // global addr lo
    g0[3] = (unsigned)((ga >> 32) & 0x01FFFFFFu) | (2u << 30); // addr[56:32] | type=2
    i32x8 g1;
    g1[0] = (int)(1u << 16);                                 // data_size=1 (2 bytes)
    g1[1] = (int)((tensor_x & 0xFFFFu) << 16);               // tensor_dim0[15:0]
    g1[2] = (int)(((tensor_x >> 16) & 0xFFFFu) | ((tensor_y & 0xFFFFu) << 16));
    g1[3] = (int)(((tensor_y >> 16) & 0xFFFFu) | ((tile_x & 0xFFFFu) << 16)); // tile_dim0
    g1[4] = (int)(tile_y & 0xFFFFu);                         // tile_dim1 (tile_dim2=0)
    g1[5] = (int)stride_x;                                   // tensor_dim0_stride lo
    g1[6] = 0;                                               // stride hi / dim1_stride lo
    g1[7] = 0;                                               // dim1_stride hi
    i32x4 z4 = {0, 0, 0, 0};
    i32x8 z8 = {0, 0, 0, 0, 0, 0, 0, 0};
    __builtin_amdgcn_tensor_load_to_lds(g0, g1, z4, z4, z8, 0);
}

// ---------------------------------------------------------------------------
// fp32 -> f16 convert
// ---------------------------------------------------------------------------
__global__ __launch_bounds__(256) void cvt_f16(const float* __restrict__ src,
                                               _Float16* __restrict__ dst, int n) {
    int i = blockIdx.x * 256 + threadIdx.x;
    if (i < n) dst[i] = (_Float16)src[i];
}

// ---------------------------------------------------------------------------
// QKV GEMM: (16384 x 768) * (768 x 2304) + bias -> f16 qkv
// block tile 128x128, 8 waves, each wave 64x32 (4 Mtiles x 2 Ntiles)
// A tile staged by the Tensor Data Mover; W tile staged transposed manually.
// ---------------------------------------------------------------------------
__global__ __launch_bounds__(256) void qkv_gemm(const _Float16* __restrict__ X,
                                                const _Float16* __restrict__ W,
                                                const float* __restrict__ bias,
                                                _Float16* __restrict__ Q) {
    __shared__ __align__(16) _Float16 As[128 * 32];   // row stride 32 (TDM-packed)
    __shared__ __align__(16) _Float16 Bt[128 * 40];

    const int bn = blockIdx.x * 128;
    const int bm = blockIdx.y * 128;
    const int tid = threadIdx.x, lane = tid & 31, wv = tid >> 5;
    const int wm = (wv >> 2) * 64, wn = (wv & 3) * 32;
    const int n15 = lane & 15;
    const int g = lane >> 4;

    v8f acc[4][2];
#pragma unroll
    for (int a = 0; a < 4; ++a)
#pragma unroll
        for (int b = 0; b < 2; ++b) acc[a][b] = v8f{};

    for (int kk = 0; kk < GK; kk += 32) {
        if (wv == 0) {
            unsigned as_off = (unsigned)(unsigned long long)(const void*)&As[0];
            tdm_load_2d(X + (size_t)bm * GK + kk, as_off,
                        /*tile*/ 32u, 128u,
                        /*tensor*/ (unsigned)(GK - kk), (unsigned)(16384 - bm),
                        /*stride*/ (unsigned)GK);
        }
#pragma unroll
        for (int i = 0; i < 16; ++i) {
            int idx = i * 256 + tid;
            int k2 = idx >> 7, n2 = idx & 127;
            Bt[n2 * 40 + k2] = W[(kk + k2) * GN + bn + n2];
        }
        if (wv == 0) __builtin_amdgcn_s_wait_tensorcnt(0);
        __syncthreads();

        v16h aF[4], bF[2];
#pragma unroll
        for (int mt = 0; mt < 4; ++mt) {
            const _Float16* p = &As[(wm + mt * 16 + n15) * 32 + g * 8];
            aF[mt] = ld16(p, p + 16);
        }
#pragma unroll
        for (int nt = 0; nt < 2; ++nt) {
            const _Float16* p = &Bt[(wn + nt * 16 + n15) * 40 + g * 8];
            bF[nt] = ld16(p, p + 16);
        }
#pragma unroll
        for (int mt = 0; mt < 4; ++mt)
#pragma unroll
            for (int nt = 0; nt < 2; ++nt)
                acc[mt][nt] = wmma_f16(aF[mt], bF[nt], acc[mt][nt]);
        __syncthreads();
    }

#pragma unroll
    for (int nt = 0; nt < 2; ++nt) {
        int col = bn + wn + nt * 16 + n15;
        float bv = bias[col];
#pragma unroll
        for (int mt = 0; mt < 4; ++mt)
#pragma unroll
            for (int v = 0; v < 8; ++v) {
                long row = bm + wm + mt * 16 + v + g * 8;
                Q[row * GN + col] = (_Float16)(acc[mt][nt][v] + bv);
            }
    }
}

// ---------------------------------------------------------------------------
// RoPE in-place on q,k halves of qkv (first 32 dims of each 64-dim head)
// one thread per (token, head, pair, q/k) : 16384*12*16*2 = 6,291,456
// ---------------------------------------------------------------------------
__global__ __launch_bounds__(256) void rope_kernel(_Float16* __restrict__ QKV) {
    int i = blockIdx.x * 256 + threadIdx.x;
    if (i >= 6291456) return;
    int qk = i & 1;
    int r = i >> 1;
    int p = r & 15; r >>= 4;
    int h = r % 12;
    int tg = r / 12;            // global token 0..16383
    int n = tg & 4095;          // position within sequence
    long base = (long)tg * GN + qk * 768 + h * 64 + 2 * p;
    // inv_freq = 10000^(-p/16) = 2^(-p * log2(10000)/16)
    float inv = exp2f(-(float)p * (13.2877123795494f / 16.0f));
    float fr = (float)n * inv;
    float sn, cs;
    __sincosf(fr, &sn, &cs);
    float x0 = (float)QKV[base], x1 = (float)QKV[base + 1];
    QKV[base]     = (_Float16)(x0 * cs - x1 * sn);
    QKV[base + 1] = (_Float16)(x1 * cs + x0 * sn);
}

// ---------------------------------------------------------------------------
// Sliding-window attention, one block per (head-seq, window).
// 8 waves; wave owns 16 query rows; flash-style over 12 KV chunks of 32.
// Q tile staged by TDM; K staged row-major, V staged transposed.
// ---------------------------------------------------------------------------
__global__ __launch_bounds__(256) void local_attn(const _Float16* __restrict__ QKV,
                                                  float* __restrict__ O) {
    __shared__ __align__(16) _Float16 Qs[128 * 64];   // row stride 64 (TDM-packed)
    __shared__ __align__(16) _Float16 Ks[384 * 72];
    __shared__ __align__(16) _Float16 Vt[64 * 392];
    __shared__ __align__(16) _Float16 Ps[8 * 16 * 32];

    const int wi = blockIdx.x;          // window 0..31
    const int bh = blockIdx.y;          // 0..47
    const int b = bh / 12, h = bh % 12;
    const int tid = threadIdx.x, lane = tid & 31, wv = tid >> 5;
    const int n15 = lane & 15;
    const int g = lane >> 4;
    const long tokQ0 = (long)b * 4096 + wi * 128;

    // ---- TDM: load Q tile (128 rows x 64 cols, row stride GN) ----
    if (wv == 0) {
        unsigned qs_off = (unsigned)(unsigned long long)(const void*)&Qs[0];
        tdm_load_2d(QKV + tokQ0 * GN + h * 64, qs_off,
                    /*tile*/ 64u, 128u, /*tensor*/ 64u, 128u, /*stride*/ (unsigned)GN);
    }
    // ---- load K tile (384x64) and V tile transposed (64x384) ----
    for (int idx = tid; idx < 24576; idx += 256) {
        int r = idx >> 6, c = idx & 63;          // r = j, c = d
        int tok = wi * 128 - 128 + r;
        _Float16 kv = (_Float16)0.f, vv = (_Float16)0.f;
        if (tok >= 0 && tok < 4096) {
            long base = ((long)b * 4096 + tok) * GN + h * 64 + c;
            kv = QKV[768 + base];
            vv = QKV[1536 + base];
        }
        Ks[r * 72 + c] = kv;
        Vt[c * 392 + r] = vv;
    }
    if (wv == 0) __builtin_amdgcn_s_wait_tensorcnt(0);
    __syncthreads();

    // ---- per-wave Q A-fragments (rows m0..m0+15, K split 0..31 / 32..63) ----
    const int m0 = wv * 16;
    v16h aQ[2];
#pragma unroll
    for (int kb = 0; kb < 2; ++kb) {
        const _Float16* p = &Qs[(m0 + n15) * 64 + kb * 32 + g * 8];
        aQ[kb] = ld16(p, p + 16);
    }

    v8f accO[4];
#pragma unroll
    for (int d = 0; d < 4; ++d) accO[d] = v8f{};
    float m_i[8], l_i[8];
#pragma unroll
    for (int v = 0; v < 8; ++v) { m_i[v] = -1e30f; l_i[v] = 0.f; }

    _Float16* pbuf = &Ps[wv * 512];

    for (int ch = 0; ch < 12; ++ch) {
        const int j0 = ch * 32;
        v8f s[2];
        // ---- S = (Q . K^T) * scale for two 16-wide column tiles ----
#pragma unroll
        for (int t = 0; t < 2; ++t) {
            v8f c = v8f{};
#pragma unroll
            for (int kb = 0; kb < 2; ++kb) {
                const _Float16* p = &Ks[(j0 + t * 16 + n15) * 72 + kb * 32 + g * 8];
                v16h bK = ld16(p, p + 16);
                c = wmma_f16(aQ[kb], bK, c);
            }
            int tok = wi * 128 - 128 + j0 + t * 16 + n15;
            bool bad = (tok < 0) || (tok >= 4096);
#pragma unroll
            for (int v = 0; v < 8; ++v)
                s[t][v] = bad ? -1e30f : c[v] * 0.125f;  // 1/sqrt(64)
        }
        // ---- online softmax (row = v + 8g, columns across 16-lane group) ----
#pragma unroll
        for (int v = 0; v < 8; ++v) {
            float mx = fmaxf(s[0][v], s[1][v]);
            mx = fmaxf(mx, __shfl_xor(mx, 1));
            mx = fmaxf(mx, __shfl_xor(mx, 2));
            mx = fmaxf(mx, __shfl_xor(mx, 4));
            mx = fmaxf(mx, __shfl_xor(mx, 8));
            float mnew = fmaxf(m_i[v], mx);
            float corr = __expf(m_i[v] - mnew);
            float p0 = __expf(s[0][v] - mnew);
            float p1 = __expf(s[1][v] - mnew);
            s[0][v] = p0; s[1][v] = p1;
            float rs = p0 + p1;
            rs += __shfl_xor(rs, 1);
            rs += __shfl_xor(rs, 2);
            rs += __shfl_xor(rs, 4);
            rs += __shfl_xor(rs, 8);
            m_i[v] = mnew;
            l_i[v] = l_i[v] * corr + rs;
#pragma unroll
            for (int d = 0; d < 4; ++d) accO[d][v] *= corr;
        }
        // ---- stage P (C-layout -> LDS row-major 16x32) ----
#pragma unroll
        for (int t = 0; t < 2; ++t)
#pragma unroll
            for (int v = 0; v < 8; ++v)
                pbuf[(v + g * 8) * 32 + t * 16 + n15] = (_Float16)s[t][v];
        __syncthreads();
        // ---- O += P (16x32) . V (32x64) ----
        {
            const _Float16* p = &pbuf[n15 * 32 + g * 8];
            v16h aP = ld16(p, p + 16);
#pragma unroll
            for (int d = 0; d < 4; ++d) {
                const _Float16* q = &Vt[(d * 16 + n15) * 392 + j0 + g * 8];
                v16h bV = ld16(q, q + 16);
                accO[d] = wmma_f16(aP, bV, accO[d]);
            }
        }
        __syncthreads();
    }

    // ---- normalize and write fp32 output ----
#pragma unroll
    for (int d = 0; d < 4; ++d)
#pragma unroll
        for (int v = 0; v < 8; ++v) {
            int row = m0 + v + g * 8;
            long tok = tokQ0 + row;
            O[tok * 768 + h * 64 + d * 16 + n15] = accO[d][v] / l_i[v];
        }
}

// ---------------------------------------------------------------------------
// launch
// ---------------------------------------------------------------------------
extern "C" void kernel_launch(void* const* d_in, const int* in_sizes, int n_in,
                              void* d_out, int out_size, void* d_ws, size_t ws_size,
                              hipStream_t stream) {
    (void)in_sizes; (void)n_in; (void)out_size; (void)ws_size;
    const float* x    = (const float*)d_in[0];   // (4,4096,768)
    const float* w    = (const float*)d_in[1];   // (768,2304)
    const float* bias = (const float*)d_in[2];   // (2304,)
    // d_in[3] = window_size (=128), baked in.

    char* ws = (char*)d_ws;
    _Float16* qkvh = (_Float16*)ws;                                   // 75,497,472 B
    _Float16* xh   = (_Float16*)(ws + 75497472L);                     // 25,165,824 B
    _Float16* wh   = (_Float16*)(ws + 75497472L + 25165824L);         //  3,538,944 B
    float* out = (float*)d_out;

    const int NX = 4 * 4096 * 768;      // 12,582,912
    const int NW = 768 * 2304;          //  1,769,472
    cvt_f16<<<(NX + 255) / 256, 256, 0, stream>>>(x, xh, NX);
    cvt_f16<<<(NW + 255) / 256, 256, 0, stream>>>(w, wh, NW);
    qkv_gemm<<<dim3(GN / 128, 16384 / 128), 256, 0, stream>>>(xh, wh, bias, qkvh);
    rope_kernel<<<(6291456 + 255) / 256, 256, 0, stream>>>(qkvh);
    local_attn<<<dim3(32, 48), 256, 0, stream>>>(qkvh, out);
}
